// GGNN_37709812859115
// MI455X (gfx1250) — matrix operand order repaired and checked by reading, not compiled, not run on hardware
//
#include <hip/hip_runtime.h>

// ---------------------------------------------------------------------------
// GGNN for MI455X (gfx1250, wave32, WMMA).
// Heavy op: per-pair MLP (512*512 rows x 128x128 x 128x64) fused into one
// WMMA kernel, ~12.9 GFLOP/step; intermediates never touch HBM.
// Round 2: B-matrices pre-packed into WMMA fragment layout in LDS
// (contiguous 32B per lane -> ds_load_b128), vectorized A-fragment
// construction, hoisted B fragments shared by both row-tiles.
// ---------------------------------------------------------------------------

typedef __attribute__((ext_vector_type(16))) _Float16 v16h;
typedef __attribute__((ext_vector_type(8)))  _Float16 v8h;
typedef __attribute__((ext_vector_type(8)))  float    v8f;
typedef __attribute__((ext_vector_type(4)))  float    v4f;

#define WMMA16(a, b, c) \
  __builtin_amdgcn_wmma_f32_16x16x32_f16(false, (a), false, (b), (short)0, (c), false, false)

#define N_NODES 512
#define S_DIM   64
#define M_DIM   64
#define HM_DIM  128
#define HR_DIM  128
#define STEPS   5

// ---------------------------------------------------------------------------
__global__ __launch_bounds__(256) void zero_f32(float* __restrict__ p, int n) {
  int i = blockIdx.x * 256 + threadIdx.x;
  if (i < n) p[i] = 0.0f;
}

// msg[j][n] = 512 * mb3[n]  (fold the Sum_i mb3 term)
__global__ __launch_bounds__(256) void init_msg(float* __restrict__ msg,
                                                const float* __restrict__ mb3) {
  int i = blockIdx.x * 256 + threadIdx.x;
  if (i < N_NODES * M_DIM) msg[i] = 512.0f * mb3[i & 63];
}

// ---------------------------------------------------------------------------
// prep: Ai[i][k] = (h@W_hi)[i][k] + b[i]*w_bi[k] + mb1[k]
//       Bj[i][k] = (h@W_hj)[i][k] + b[i]*w_bj[k]
// mW1 rows: [0..63]=W_hi, [64..127]=W_hj, 128=w_J, 129=w_bi, 130=w_bj
// ---------------------------------------------------------------------------
__global__ __launch_bounds__(256) void prep_kernel(const float* __restrict__ h,
                                                   const float* __restrict__ mW1,
                                                   const float* __restrict__ mb1,
                                                   const float* __restrict__ b,
                                                   float* __restrict__ Ai,
                                                   float* __restrict__ Bj) {
  int idx = blockIdx.x * 256 + threadIdx.x;   // 512*128
  if (idx >= N_NODES * HM_DIM) return;
  int i = idx >> 7;
  int k = idx & 127;
  float bi = b[i];
  float ai = mb1[k] + bi * mW1[129 * HM_DIM + k];
  float bj = bi * mW1[130 * HM_DIM + k];
  #pragma unroll 8
  for (int s = 0; s < S_DIM; ++s) {
    float hv = h[i * S_DIM + s];
    ai += hv * mW1[s * HM_DIM + k];
    bj += hv * mW1[(S_DIM + s) * HM_DIM + k];
  }
  Ai[idx] = ai;
  Bj[idx] = bj;
}

// ---------------------------------------------------------------------------
// Fused pair-MLP + message reduction.
// grid = (32 j-tiles, 8 i-chunks), block = 256 threads = 8 waves.
// Subtile rows: p = ii*16 + jj  (row-tile rt <-> ii, in-tile row <-> jj).
// B-fragment layout (packed): element t of lane l = W[kofs + (t&7) + (t>>3)*16][n] with
//   n = l&15, kofs = kb*32 + (l>>4)*8   (per CDNA5 ISA B 32x16 layout)
// A-fragment layout: lane l covers row m=l&15, halves 0..7 -> K=base..base+7,
//   halves 8..15 -> K=base+16..base+23, base = kb*32 + (l>>4)*8.
// ---------------------------------------------------------------------------
__global__ __launch_bounds__(256) void pair_msg_kernel(
    const float* __restrict__ J,   const float* __restrict__ mW1,
    const float* __restrict__ mW2, const float* __restrict__ mb2,
    const float* __restrict__ mW3,
    const float* __restrict__ Ai,  const float* __restrict__ Bj,
    float* __restrict__ msg) {
  constexpr int YS  = 136;   // halfs per row of sY (16B aligned, bank-rotating)
  constexpr int BJS = 132;   // floats per row of sBj

  __shared__ v16h sW2f[32 * 32];                 // 32 frags (ct*4+kb) x 32 lanes, 32,768 B
  __shared__ v16h sW3f[16 * 32];                 // 16 frags, 16,384 B
  __shared__ __attribute__((aligned(32))) _Float16 sY[256 * YS];  // 69,632 B
  __shared__ __attribute__((aligned(16))) float sAi[16 * HM_DIM]; //  8,192 B
  __shared__ __attribute__((aligned(16))) float sBj[16 * BJS];    //  8,448 B
  __shared__ float sJ[16 * 16];                  // 1,024 B
  __shared__ __attribute__((aligned(16))) float swJ[HM_DIM];      // 512 B
  __shared__ float smb2[HM_DIM];                 // 512 B
  __shared__ float sMsg[16 * M_DIM];             // 4,096 B

  const int tid    = threadIdx.x;
  const int lane   = tid & 31;
  const int wave   = tid >> 5;
  const int jbase  = blockIdx.x * 16;
  const int ichunk = blockIdx.y;                 // 8 chunks x 64 i's

  // ---- pack weights into B-fragment layout ---------------------------
  {
    _Float16* w2 = (_Float16*)sW2f;
    for (int idx = tid; idx < HM_DIM * HM_DIM; idx += 256) {
      int k = idx >> 7, n = idx & 127;
      int kb = k >> 5, kr = k & 31;
      int h2 = (kr >> 3) & 1, t = (kr & 7) | ((kr >> 4) << 3);
      int ct = n >> 4, nn = n & 15;
      w2[(((ct * 4 + kb) * 32) + (h2 * 16 + nn)) * 16 + t] = (_Float16)mW2[idx];
    }
    _Float16* w3 = (_Float16*)sW3f;
    for (int idx = tid; idx < HM_DIM * M_DIM; idx += 256) {
      int k = idx >> 6, n = idx & 63;
      int kb = k >> 5, kr = k & 31;
      int h2 = (kr >> 3) & 1, t = (kr & 7) | ((kr >> 4) << 3);
      int ct = n >> 4, nn = n & 15;
      w3[(((ct * 4 + kb) * 32) + (h2 * 16 + nn)) * 16 + t] = (_Float16)mW3[idx];
    }
  }
  for (int idx = tid; idx < 16 * HM_DIM; idx += 256) {
    int jj = idx >> 7, k = idx & 127;
    sBj[jj * BJS + k] = Bj[(jbase + jj) * HM_DIM + k];
  }
  for (int idx = tid; idx < HM_DIM; idx += 256) {
    swJ[idx]  = mW1[128 * HM_DIM + idx];    // w_J row
    smb2[idx] = mb2[idx];
  }
  for (int idx = tid; idx < 16 * M_DIM; idx += 256) sMsg[idx] = 0.0f;

  const int jj    = lane & 15;
  const int hi2   = lane >> 4;   // 0/1 half of wave
  const int kb8   = hi2 * 8;     // A-frag K base   (lanes>=16: +8)
  const int mbase = hi2 * 8;     // C/D row base    (lanes>=16: M+8)

  // persistent GEMM2 accumulators: [row-tile pair][col-tile], Sum over all i
  v8f acc2[2][4];
  #pragma unroll
  for (int a = 0; a < 2; ++a)
    #pragma unroll
    for (int c = 0; c < 4; ++c) acc2[a][c] = (v8f){0, 0, 0, 0, 0, 0, 0, 0};

  #pragma unroll 1
  for (int it = 0; it < 4; ++it) {
    const int ibase = ichunk * 64 + it * 16;
    __syncthreads();   // prev iter done reading sAi/sJ/sY
    for (int idx = tid; idx < 16 * HM_DIM; idx += 256) {
      int ii = idx >> 7, k = idx & 127;
      sAi[ii * HM_DIM + k] = Ai[(ibase + ii) * HM_DIM + k];
    }
    {
      int ii = tid >> 4, j2 = tid & 15;   // 256 threads <-> 16x16 tile
      sJ[tid] = J[(ibase + ii) * N_NODES + jbase + j2];
    }
    if (it < 3) {  // prefetch next subtile's Ai rows (global_prefetch_b8)
      __builtin_prefetch(Ai + (ibase + 16) * HM_DIM + tid * 8, 0, 3);
    }
    __syncthreads();

    // ---- build h1 A-fragments for both row-tiles ---------------------
    v16h afr[2][4];
    float jvv[2];
    #pragma unroll
    for (int rt2 = 0; rt2 < 2; ++rt2) {
      const int rt = wave * 2 + rt2;          // ii = rt
      jvv[rt2] = sJ[rt * 16 + jj];
      const float jv = jvv[rt2];
      const v4f* pAi = (const v4f*)(sAi + rt * HM_DIM);
      const v4f* pBj = (const v4f*)(sBj + jj * BJS);
      const v4f* pwJ = (const v4f*)swJ;
      #pragma unroll
      for (int kb = 0; kb < 4; ++kb) {
        const int q = (kb * 32 + kb8) >> 2;   // v4f index; +16 floats -> q+4
        v4f a0 = pAi[q], a1 = pAi[q + 1], a2 = pAi[q + 4], a3 = pAi[q + 5];
        v4f b0 = pBj[q], b1 = pBj[q + 1], b2 = pBj[q + 4], b3 = pBj[q + 5];
        v4f w0 = pwJ[q], w1 = pwJ[q + 1], w2 = pwJ[q + 4], w3 = pwJ[q + 5];
        v16h a;
        #pragma unroll
        for (int e = 0; e < 4; ++e) {
          a[e]      = (_Float16)fmaxf(a0[e] + b0[e] + jv * w0[e], 0.0f);
          a[4 + e]  = (_Float16)fmaxf(a1[e] + b1[e] + jv * w1[e], 0.0f);
          a[8 + e]  = (_Float16)fmaxf(a2[e] + b2[e] + jv * w2[e], 0.0f);
          a[12 + e] = (_Float16)fmaxf(a3[e] + b3[e] + jv * w3[e], 0.0f);
        }
        afr[rt2][kb] = a;
      }
    }

    // ---- GEMM1: h1 @ mW2, +mb2, relu -> sY (f16) ---------------------
    #pragma unroll
    for (int ct = 0; ct < 8; ++ct) {
      v8f acc0 = (v8f){0, 0, 0, 0, 0, 0, 0, 0};
      v8f acc1 = (v8f){0, 0, 0, 0, 0, 0, 0, 0};
      #pragma unroll
      for (int kb = 0; kb < 4; ++kb) {
        v16h bfr = sW2f[(ct * 4 + kb) * 32 + lane];   // 2x ds_load_b128
        acc0 = WMMA16(afr[0][kb], bfr, acc0);
        acc1 = WMMA16(afr[1][kb], bfr, acc1);
      }
      const int n = ct * 16 + jj;
      const float bias = smb2[n];
      const int r0 = ((wave * 2 + 0) * 16 + mbase) * YS + n;
      const int r1 = ((wave * 2 + 1) * 16 + mbase) * YS + n;
      #pragma unroll
      for (int r = 0; r < 8; ++r) {
        sY[r0 + r * YS] = (_Float16)fmaxf(acc0[r] + bias, 0.0f);
        sY[r1 + r * YS] = (_Float16)fmaxf(acc1[r] + bias, 0.0f);
      }
    }
    __syncthreads();

    // ---- GEMM2: h2 @ mW3, accumulate over i in registers -------------
    v16h afr2[2][4];
    #pragma unroll
    for (int rt2 = 0; rt2 < 2; ++rt2) {
      const int row = ((wave * 2 + rt2) * 16 + jj) * YS;
      #pragma unroll
      for (int kb = 0; kb < 4; ++kb) {
        const int base = kb * 32 + kb8;
        v8h lo = *(const v8h*)(sY + row + base);        // ds_load_b128
        v8h hh = *(const v8h*)(sY + row + base + 16);   // ds_load_b128
        afr2[rt2][kb] = __builtin_shufflevector(
            lo, hh, 0, 1, 2, 3, 4, 5, 6, 7, 8, 9, 10, 11, 12, 13, 14, 15);
      }
    }
    #pragma unroll
    for (int ct = 0; ct < 4; ++ct) {
      #pragma unroll
      for (int kb = 0; kb < 4; ++kb) {
        v16h bfr = sW3f[(ct * 4 + kb) * 32 + lane];
        acc2[0][ct] = WMMA16(afr2[0][kb], bfr, acc2[0][ct]);
        acc2[1][ct] = WMMA16(afr2[1][kb], bfr, acc2[1][ct]);
      }
    }
  }

  // ---- reduce acc2 across waves -> sMsg -> global msg ----------------
  __syncthreads();
  #pragma unroll
  for (int ct = 0; ct < 4; ++ct) {
    const int n = ct * 16 + jj;
    #pragma unroll
    for (int r = 0; r < 8; ++r) {
      float v = acc2[0][ct][r] + acc2[1][ct][r];   // same (jj,n) for both rt
      atomicAdd(&sMsg[(mbase + r) * M_DIM + n], v);
    }
  }
  __syncthreads();
  for (int idx = tid; idx < 16 * M_DIM; idx += 256) {
    int j2 = idx >> 6, n = idx & 63;
    atomicAdd(&msg[(jbase + j2) * M_DIM + n], sMsg[idx]);
  }
}

// ---------------------------------------------------------------------------
// GRU update: h_new = gru(msg, h)
// ---------------------------------------------------------------------------
__device__ __forceinline__ float sigm(float x) { return 1.0f / (1.0f + __expf(-x)); }

__global__ __launch_bounds__(256) void gru_kernel(const float* __restrict__ msg,
                                                  const float* __restrict__ h,
                                                  const float* __restrict__ W_ih,
                                                  const float* __restrict__ b_ih,
                                                  const float* __restrict__ W_hh,
                                                  const float* __restrict__ b_hh,
                                                  float* __restrict__ hn) {
  int idx = blockIdx.x * 256 + threadIdx.x;   // 512*64
  if (idx >= N_NODES * S_DIM) return;
  int i = idx >> 6;
  int s = idx & 63;
  float gxr = b_ih[s], gxz = b_ih[64 + s], gxn = b_ih[128 + s];
  float ghr = b_hh[s], ghz = b_hh[64 + s], ghn = b_hh[128 + s];
  #pragma unroll 4
  for (int m = 0; m < M_DIM; ++m) {
    float mv = msg[i * M_DIM + m];
    gxr += mv * W_ih[s * M_DIM + m];
    gxz += mv * W_ih[(64 + s) * M_DIM + m];
    gxn += mv * W_ih[(128 + s) * M_DIM + m];
    float hv = h[i * S_DIM + m];
    ghr += hv * W_hh[s * S_DIM + m];
    ghz += hv * W_hh[(64 + s) * S_DIM + m];
    ghn += hv * W_hh[(128 + s) * S_DIM + m];
  }
  float r  = sigm(gxr + ghr);
  float z  = sigm(gxz + ghz);
  float nc = tanhf(gxn + r * ghn);
  hn[idx] = (1.0f - z) * nc + z * h[idx];
}

// ---------------------------------------------------------------------------
// Readout MLP + 2-class softmax. One block (128 thr) per node.
// ---------------------------------------------------------------------------
__global__ __launch_bounds__(128) void readout_kernel(
    const float* __restrict__ h,
    const float* __restrict__ rW1, const float* __restrict__ rb1,
    const float* __restrict__ rW2, const float* __restrict__ rb2,
    const float* __restrict__ rW3, const float* __restrict__ rb3,
    float* __restrict__ out) {
  __shared__ float s1[HR_DIM];
  __shared__ float s2[HR_DIM];
  __shared__ float so[2];
  const int i = blockIdx.x;
  const int k = threadIdx.x;

  float o1 = rb1[k];
  #pragma unroll 8
  for (int s = 0; s < S_DIM; ++s) o1 += h[i * S_DIM + s] * rW1[s * HR_DIM + k];
  s1[k] = fmaxf(o1, 0.0f);
  __syncthreads();

  float o2 = rb2[k];
  #pragma unroll 8
  for (int m = 0; m < HR_DIM; ++m) o2 += s1[m] * rW2[m * HR_DIM + k];
  s2[k] = fmaxf(o2, 0.0f);
  __syncthreads();

  if (k < 2) {
    float o3 = rb3[k];
    for (int m = 0; m < HR_DIM; ++m) o3 += s2[m] * rW3[m * 2 + k];
    so[k] = o3;
  }
  __syncthreads();
  if (k == 0) {
    float mx = fmaxf(so[0], so[1]);
    float e0 = __expf(so[0] - mx);
    float e1 = __expf(so[1] - mx);
    float inv = 1.0f / (e0 + e1);
    out[i * 2 + 0] = e0 * inv;
    out[i * 2 + 1] = e1 * inv;
  }
}

// ---------------------------------------------------------------------------
// host launcher
// ---------------------------------------------------------------------------
extern "C" void kernel_launch(void* const* d_in, const int* in_sizes, int n_in,
                              void* d_out, int out_size, void* d_ws, size_t ws_size,
                              hipStream_t stream) {
  const float* J    = (const float*)d_in[0];
  const float* b    = (const float*)d_in[1];
  const float* mW1  = (const float*)d_in[2];
  const float* mb1  = (const float*)d_in[3];
  const float* mW2  = (const float*)d_in[4];
  const float* mb2  = (const float*)d_in[5];
  const float* mW3  = (const float*)d_in[6];
  const float* mb3  = (const float*)d_in[7];
  const float* W_ih = (const float*)d_in[8];
  const float* b_ih = (const float*)d_in[9];
  const float* W_hh = (const float*)d_in[10];
  const float* b_hh = (const float*)d_in[11];
  const float* rW1  = (const float*)d_in[12];
  const float* rb1  = (const float*)d_in[13];
  const float* rW2  = (const float*)d_in[14];
  const float* rb2  = (const float*)d_in[15];
  const float* rW3  = (const float*)d_in[16];
  const float* rb3  = (const float*)d_in[17];

  char* ws = (char*)d_ws;
  float* h0  = (float*)(ws + 0);        // 512*64*4   = 131072
  float* h1  = (float*)(ws + 131072);   // 512*64*4
  float* Ai  = (float*)(ws + 262144);   // 512*128*4  = 262144
  float* Bj  = (float*)(ws + 524288);   // 512*128*4
  float* msg = (float*)(ws + 786432);   // 512*64*4

  zero_f32<<<128, 256, 0, stream>>>(h0, N_NODES * S_DIM);

  float* hc = h0;
  float* hn = h1;
  for (int step = 0; step < STEPS; ++step) {
    prep_kernel<<<256, 256, 0, stream>>>(hc, mW1, mb1, b, Ai, Bj);
    init_msg<<<128, 256, 0, stream>>>(msg, mb3);
    pair_msg_kernel<<<dim3(32, 8), 256, 0, stream>>>(J, mW1, mW2, mb2, mW3,
                                                     Ai, Bj, msg);
    gru_kernel<<<128, 256, 0, stream>>>(msg, hc, W_ih, b_ih, W_hh, b_hh, hn);
    float* t = hc; hc = hn; hn = t;
  }
  readout_kernel<<<N_NODES, 128, 0, stream>>>(hc, rW1, rb1, rW2, rb2, rW3, rb3,
                                              (float*)d_out);
}
